// GPS_14525579395561
// MI455X (gfx1250) — matrix operand order repaired and checked by reading, not compiled, not run on hardware
//
#include <hip/hip_runtime.h>
#include <hip/hip_bf16.h>

typedef __attribute__((ext_vector_type(16))) _Float16 v16h;
typedef __attribute__((ext_vector_type(8)))  _Float16 v8h;
typedef __attribute__((ext_vector_type(8)))  float    v8f;
typedef int b128_t __attribute__((vector_size(16)));  // matches builtin param type

#define N_NODES 32768
#define G_GRAPHS 64
#define NG 512
#define E_EDGES 524288
#define F_IN 16
#define C_CH 128
#define L_LAYERS 3
#define H_HEADS 4
#define DH_DIM 32
#define BN_EPS 1e-5f

// Use the gfx1250 async global->LDS path when the toolchain exposes it.
#if defined(__AMDGCN__) && defined(__has_builtin)
#  if __has_builtin(__builtin_amdgcn_global_load_async_to_lds_b128)
#    define USE_ASYNC_LDS 1
#  endif
#endif
#ifndef USE_ASYNC_LDS
#  define USE_ASYNC_LDS 0
#endif

__device__ __forceinline__ void wait_async_lds() {
#if defined(__AMDGCN__)
#  if defined(__has_builtin) && __has_builtin(__builtin_amdgcn_s_wait_asynccnt)
  __builtin_amdgcn_s_wait_asynccnt(0);
#  else
  asm volatile("s_wait_asynccnt 0" ::: "memory");
#  endif
#endif
}

// ---------------------------------------------------------------------------
// A-fragment loader for v_wmma_f32_16x16x32_f16 (16x32 f16 A, row-major src).
// lane m = lane&15 holds row m; sel = lane>>4 selects K sub-chunk:
//   elems 0..15 = A[m][sel*16 .. sel*16+15]
// ---------------------------------------------------------------------------
__device__ __forceinline__ v16h load_a_frag(const _Float16* p, int sel) {
  v8h a0 = *(const v8h*)(p + sel * 16);
  v8h a1 = *(const v8h*)(p + sel * 16 + 8);
  return __builtin_shufflevector(a0, a1, 0, 1, 2, 3, 4, 5, 6, 7,
                                 8, 9, 10, 11, 12, 13, 14, 15);
}

// ---------------------------------------------------------------------------
// Fused WMMA GEMM:  out = epilogue( A[M,K](f16) x W[Nout,K]^T(f16) )
// Block = 128 threads (4 waves), block tile = 128 rows x 64 cols.
// The 64xK weight slab is staged into LDS via GLOBAL_LOAD_ASYNC_TO_LDS_B128
// (ASYNCcnt / s_wait_asynccnt) when available; B-fragments then come from LDS.
// Each wave register-blocks 32x64: 2 A-frags x 4 B-frags -> 8 wmma per k-step.
// Epilogue: +bias, optional ReLU, optional f32 residual, optional eval-BN.
// Grid = (Nout/64, M/128).
// ---------------------------------------------------------------------------
__global__ __launch_bounds__(128) void gemm_wmma_kernel(
    const _Float16* __restrict__ A, const _Float16* __restrict__ W,
    const float* __restrict__ bias, const float* __restrict__ res,
    const float* __restrict__ bn_g, const float* __restrict__ bn_b,
    const float* __restrict__ bn_m, const float* __restrict__ bn_v,
    float* __restrict__ out32, _Float16* __restrict__ out16,
    int K, int Nout, int relu_flag)
{
  __shared__ __align__(16) _Float16 lds_w[64 * 256];  // 32 KB max (K<=256)

  const int tid = threadIdx.x;
  const int col0 = blockIdx.x * 64;
  const _Float16* Wslab = W + (size_t)col0 * K;
  const int slab_halves = 64 * K;  // multiple of 1024 -> uniform loop

#if USE_ASYNC_LDS
  for (int i = tid * 8; i < slab_halves; i += 128 * 8) {
    __builtin_amdgcn_global_load_async_to_lds_b128(
        (__attribute__((address_space(1))) b128_t*)(Wslab + i),
        (__attribute__((address_space(3))) b128_t*)(lds_w + i), 0, 0);
  }
  wait_async_lds();
  __syncthreads();
#else
  for (int i = tid * 8; i < slab_halves; i += 128 * 8)
    *(v8h*)(lds_w + i) = *(const v8h*)(Wslab + i);
  __syncthreads();
#endif

  const int wave = tid >> 5;
  const int lane = tid & 31;
  const int m = lane & 15;
  const int sel = lane >> 4;
  const int rowbase = blockIdx.y * 128 + wave * 32;

  const _Float16* Arow0 = A + (size_t)(rowbase + m) * K;
  const _Float16* Arow1 = A + (size_t)(rowbase + 16 + m) * K;

  v8f zero = {};
  v8f acc[8];  // [mt*4 + nt]
#pragma unroll
  for (int i = 0; i < 8; ++i) acc[i] = zero;

  for (int k = 0; k < K; k += 32) {
    __builtin_prefetch(Arow0 + k + 256, 0, 0);  // global_prefetch_b8
    v16h a0 = load_a_frag(Arow0 + k, sel);
    v16h a1 = load_a_frag(Arow1 + k, sel);
#pragma unroll
    for (int nt = 0; nt < 4; ++nt) {
      const _Float16* wp = lds_w + (size_t)(nt * 16 + m) * K + k + sel * 16;
      v16h b = *(const v16h*)wp;
      acc[nt] = __builtin_amdgcn_wmma_f32_16x16x32_f16(
          false, a0, false, b, (short)0, acc[nt], false, false);
      acc[4 + nt] = __builtin_amdgcn_wmma_f32_16x16x32_f16(
          false, a1, false, b, (short)0, acc[4 + nt], false, false);
    }
  }

#pragma unroll
  for (int nt = 0; nt < 4; ++nt) {
    const int col = col0 + nt * 16 + m;
    const float bi = bias ? bias[col] : 0.0f;
    float s = 1.0f, t = 0.0f;
    if (bn_g) {
      const float gg = bn_g[col], bb = bn_b[col], mu = bn_m[col], vv = bn_v[col];
      s = gg * rsqrtf(vv + BN_EPS);
      t = bb - mu * s;
    }
#pragma unroll
    for (int mt = 0; mt < 2; ++mt) {
#pragma unroll
      for (int v = 0; v < 8; ++v) {
        const int row = rowbase + mt * 16 + v + sel * 8;
        float val = acc[mt * 4 + nt][v] + bi;
        if (relu_flag) val = fmaxf(val, 0.0f);
        if (res) val += res[(size_t)row * Nout + col];
        if (bn_g) val = val * s + t;
        if (out32) out32[(size_t)row * Nout + col] = val;
        if (out16) out16[(size_t)row * Nout + col] = (_Float16)val;
      }
    }
  }
}

// ---------------------------------------------------------------------------
// Fused attention: one workgroup (8 waves) per (graph g, head hh, 16 q-rows).
// S = (Q K^T)*scale in registers (4 tiles/wave), softmax via shfl + LDS,
// P stored f16 in LDS, O = P V via WMMA with K-split across waves.
// ---------------------------------------------------------------------------
__global__ __launch_bounds__(256) void attn_kernel(
    const _Float16* __restrict__ qkv, _Float16* __restrict__ o16)
{
  __shared__ __align__(16) _Float16 Pbuf[16 * 512];   // 16 KB
  __shared__ __align__(16) _Float16 Vt[32 * 512];     // 32 KB (V transposed)
  __shared__ float Opart[8 * 256];                    // 8 KB
  __shared__ float pmax[8 * 16];
  __shared__ float psum[8 * 16];
  __shared__ float rowmax[16];
  __shared__ float rowsum[16];

  const int tid = threadIdx.x;
  const int bid = blockIdx.x;
  const int qc = bid & 31;
  const int hh = (bid >> 5) & 3;
  const int g = bid >> 7;
  const int nodeBase = g * NG;
  const float scale = 0.17677669529663687f;  // 1/sqrt(32)

  // stage V^T[32][512] into LDS
  for (int i = tid; i < 32 * 512; i += 256) {
    const int d = i >> 9, node = i & 511;
    Vt[d * 512 + node] =
        qkv[(size_t)(nodeBase + node) * 384 + 256 + hh * 32 + d];
  }

  const int wave = tid >> 5;
  const int lane = tid & 31;
  const int m = lane & 15;
  const int sel = lane >> 4;

  // ---- scores: wave handles column tiles ct = wave*4 + j (K-dim = DH = 32)
  const _Float16* Qrow = qkv + (size_t)(nodeBase + qc * 16 + m) * 384 + hh * 32;
  const v16h aq = load_a_frag(Qrow, sel);
  v8f accT[4];
#pragma unroll
  for (int j = 0; j < 4; ++j) {
    const int ct = wave * 4 + j;
    const _Float16* Krow =
        qkv + (size_t)(nodeBase + ct * 16 + m) * 384 + 128 + hh * 32;
    v16h b = *(const v16h*)(Krow + sel * 16);
    v8f z = {};
    accT[j] = __builtin_amdgcn_wmma_f32_16x16x32_f16(
        false, aq, false, b, (short)0, z, false, false);
  }
#pragma unroll
  for (int j = 0; j < 4; ++j)
#pragma unroll
    for (int v = 0; v < 8; ++v) accT[j][v] *= scale;

  // ---- row max (across this wave's 64 cols, then across waves)
#pragma unroll
  for (int v = 0; v < 8; ++v) {
    float mx = fmaxf(fmaxf(accT[0][v], accT[1][v]),
                     fmaxf(accT[2][v], accT[3][v]));
#pragma unroll
    for (int off = 8; off >= 1; off >>= 1)
      mx = fmaxf(mx, __shfl_xor(mx, off, 16));
    if (m == 0) pmax[wave * 16 + sel * 8 + v] = mx;
  }
  __syncthreads();
  if (tid < 16) {
    float mv = -1e30f;
    for (int w = 0; w < 8; ++w) mv = fmaxf(mv, pmax[w * 16 + tid]);
    rowmax[tid] = mv;
  }
  __syncthreads();

  // ---- exp, store P (f16), row sums
  float sums[8];
#pragma unroll
  for (int v = 0; v < 8; ++v) sums[v] = 0.0f;
#pragma unroll
  for (int j = 0; j < 4; ++j) {
    const int colb = (wave * 4 + j) * 16 + m;
#pragma unroll
    for (int v = 0; v < 8; ++v) {
      const int r = sel * 8 + v;
      const float pv = __expf(accT[j][v] - rowmax[r]);
      Pbuf[r * 512 + colb] = (_Float16)pv;
      sums[v] += pv;
    }
  }
#pragma unroll
  for (int v = 0; v < 8; ++v) {
    float sm = sums[v];
#pragma unroll
    for (int off = 8; off >= 1; off >>= 1) sm += __shfl_xor(sm, off, 16);
    if (m == 0) psum[wave * 16 + sel * 8 + v] = sm;
  }
  __syncthreads();
  if (tid < 16) {
    float sv = 0.0f;
    for (int w = 0; w < 8; ++w) sv += psum[w * 16 + tid];
    rowsum[tid] = sv;
  }
  __syncthreads();

  // ---- O = P @ V : wave -> (nt = output 16-col tile, kq = 128-wide K slice)
  const int nt = wave & 1;
  const int kq = wave >> 1;
  const _Float16* Prow = Pbuf + m * 512 + kq * 128;
  const _Float16* Vrow = Vt + (nt * 16 + m) * 512 + kq * 128;
  v8f acc = {};
#pragma unroll
  for (int k = 0; k < 128; k += 32) {
    v16h a = load_a_frag(Prow + k, sel);
    v16h b = *(const v16h*)(Vrow + k + sel * 16);
    acc = __builtin_amdgcn_wmma_f32_16x16x32_f16(
        false, a, false, b, (short)0, acc, false, false);
  }
#pragma unroll
  for (int v = 0; v < 8; ++v)
    Opart[wave * 256 + (v + sel * 8) * 16 + m] = acc[v];
  __syncthreads();

  // ---- reduce K-slices, normalize by row sum, write O (f16, [N,128] layout)
  for (int e = tid; e < 512; e += 256) {
    const int r = e >> 5, d = e & 31;
    const int nt2 = d >> 4, n2 = d & 15;
    float sv = 0.0f;
#pragma unroll
    for (int k2 = 0; k2 < 4; ++k2)
      sv += Opart[(k2 * 2 + nt2) * 256 + r * 16 + n2];
    sv /= rowsum[r];
    o16[(size_t)(nodeBase + qc * 16 + r) * 128 + hh * 32 + d] = (_Float16)sv;
  }
}

// ---------------------------------------------------------------------------
// GINE message + scatter-add: agg[dst] += relu(h[src] + ea*w + b)
// one thread per (edge, 4-channel chunk)
// ---------------------------------------------------------------------------
__global__ __launch_bounds__(256) void gine_msg_kernel(
    const float* __restrict__ h32, const int* __restrict__ ei,
    const float* __restrict__ ea, const float* __restrict__ lew,
    const float* __restrict__ leb, float* __restrict__ agg)
{
  const size_t tid = (size_t)blockIdx.x * 256 + threadIdx.x;
  const int e = (int)(tid >> 5);
  const int c = ((int)tid & 31) << 2;
  const int src = ei[e];
  const int dst = ei[E_EDGES + e];
  const float a = ea[e];
  const float4 hv = *(const float4*)(h32 + (size_t)src * C_CH + c);
  const float4 w = *(const float4*)(lew + c);
  const float4 b = *(const float4*)(leb + c);
  float* ap = agg + (size_t)dst * C_CH + c;
  atomicAdd(ap + 0, fmaxf(hv.x + a * w.x + b.x, 0.0f));
  atomicAdd(ap + 1, fmaxf(hv.y + a * w.y + b.y, 0.0f));
  atomicAdd(ap + 2, fmaxf(hv.z + a * w.z + b.z, 0.0f));
  atomicAdd(ap + 3, fmaxf(hv.w + a * w.w + b.w, 0.0f));
}

// ---------------------------------------------------------------------------
// small helper kernels
// ---------------------------------------------------------------------------
__global__ __launch_bounds__(256) void zero_kernel(float* __restrict__ p, int n) {
  const int i = blockIdx.x * 256 + threadIdx.x;
  if (i < n) p[i] = 0.0f;
}

__global__ __launch_bounds__(256) void cvt_kernel(const float* __restrict__ a,
                                                  _Float16* __restrict__ o, int n) {
  const int i = blockIdx.x * 256 + threadIdx.x;
  if (i < n) o[i] = (_Float16)a[i];
}

__global__ __launch_bounds__(256) void addcvt_kernel(
    const float* __restrict__ a, const float* __restrict__ b,
    float* __restrict__ o32, _Float16* __restrict__ o16, int n) {
  const int i = blockIdx.x * 256 + threadIdx.x;
  if (i < n) {
    const float v = a[i] + b[i];
    o32[i] = v;
    o16[i] = (_Float16)v;
  }
}

__global__ __launch_bounds__(256) void node_emb_kernel(
    const float* __restrict__ x, const float* __restrict__ w,
    const float* __restrict__ b, float* __restrict__ h32,
    _Float16* __restrict__ h16) {
  const int i = blockIdx.x * 256 + threadIdx.x;  // N*C threads
  const int n = i >> 7, c = i & 127;
  const float* xr = x + (size_t)n * F_IN;
  const float* wr = w + (size_t)c * F_IN;
  float s = b[c];
#pragma unroll
  for (int f = 0; f < F_IN; ++f) s += xr[f] * wr[f];
  h32[i] = s;
  h16[i] = (_Float16)s;
}

__global__ __launch_bounds__(128) void pool_kernel(const float* __restrict__ h32,
                                                   float* __restrict__ gp) {
  const int g = blockIdx.x, c = threadIdx.x;
  float s = 0.0f;
  for (int i = 0; i < NG; ++i)
    s += h32[(size_t)(g * NG + i) * C_CH + c];
  gp[g * C_CH + c] = s;
}

__global__ __launch_bounds__(64) void head_kernel(
    const float* __restrict__ gp, const float* __restrict__ w1,
    const float* __restrict__ b1, const float* __restrict__ w2,
    const float* __restrict__ b2, const float* __restrict__ w3,
    const float* __restrict__ b3, float* __restrict__ out) {
  __shared__ float s1[64];
  __shared__ float s2[32];
  const int g = blockIdx.x, t = threadIdx.x;
  const float* gr = gp + g * C_CH;
  {
    float s = b1[t];
    for (int k = 0; k < 128; ++k) s += gr[k] * w1[t * 128 + k];
    s1[t] = fmaxf(s, 0.0f);
  }
  __syncthreads();
  if (t < 32) {
    float s = b2[t];
    for (int k = 0; k < 64; ++k) s += s1[k] * w2[t * 64 + k];
    s2[t] = fmaxf(s, 0.0f);
  }
  __syncthreads();
  if (t == 0) {
    float s = b3[0];
    for (int k = 0; k < 32; ++k) s += s2[k] * w3[k];
    out[g] = s;
  }
}

// ---------------------------------------------------------------------------
// orchestration
// ---------------------------------------------------------------------------
extern "C" void kernel_launch(void* const* d_in, const int* in_sizes, int n_in,
                              void* d_out, int out_size, void* d_ws, size_t ws_size,
                              hipStream_t stream) {
  (void)in_sizes; (void)n_in; (void)out_size; (void)ws_size;

  const float* x          = (const float*)d_in[0];
  const int*   edge_index = (const int*)d_in[1];
  const float* edge_attr  = (const float*)d_in[2];
  /* d_in[3] batch unused: graphs are contiguous NG-sized blocks */
  const float* node_emb_w = (const float*)d_in[4];
  const float* node_emb_b = (const float*)d_in[5];
  const float* lin_edge_w = (const float*)d_in[6];
  const float* lin_edge_b = (const float*)d_in[7];
  const float* gine_w1    = (const float*)d_in[8];
  const float* gine_b1    = (const float*)d_in[9];
  const float* gine_w2    = (const float*)d_in[10];
  const float* gine_b2    = (const float*)d_in[11];
  const float* attn_in_w  = (const float*)d_in[12];
  const float* attn_in_b  = (const float*)d_in[13];
  const float* attn_out_w = (const float*)d_in[14];
  const float* attn_out_b = (const float*)d_in[15];
  const float* norm_gamma = (const float*)d_in[16];
  const float* norm_beta  = (const float*)d_in[17];
  const float* norm_mean  = (const float*)d_in[18];
  const float* norm_var   = (const float*)d_in[19];
  const float* mlp_w1     = (const float*)d_in[20];
  const float* mlp_b1     = (const float*)d_in[21];
  const float* mlp_w2     = (const float*)d_in[22];
  const float* mlp_b2     = (const float*)d_in[23];
  const float* head_w1    = (const float*)d_in[24];
  const float* head_b1    = (const float*)d_in[25];
  const float* head_w2    = (const float*)d_in[26];
  const float* head_b2    = (const float*)d_in[27];
  const float* head_w3    = (const float*)d_in[28];
  const float* head_b3    = (const float*)d_in[29];
  float* out = (float*)d_out;

  char* p = (char*)d_ws;
  auto take = [&](size_t bytes) -> char* {
    char* r = p;
    p += (bytes + 255) & ~(size_t)255;
    return r;
  };
  const size_t NC = (size_t)N_NODES * C_CH;
  float*    h32    = (float*)take(NC * 4);
  _Float16* h16    = (_Float16*)take(NC * 2);
  float*    agg    = (float*)take(NC * 4);
  float*    loc32  = (float*)take(NC * 4);
  _Float16* loc16  = (_Float16*)take(NC * 2);
  float*    att32  = (float*)take(NC * 4);
  float*    comb32 = (float*)take(NC * 4);
  _Float16* comb16 = (_Float16*)take(NC * 2);
  _Float16* t16    = (_Float16*)take((size_t)N_NODES * 256 * 2);
  _Float16* qkv16  = (_Float16*)take((size_t)N_NODES * 384 * 2);
  _Float16* o16    = (_Float16*)take(NC * 2);
  float*    gpool  = (float*)take((size_t)G_GRAPHS * C_CH * 4);
  _Float16* wg1_16 = (_Float16*)take((size_t)3 * 16384 * 2);
  _Float16* wg2_16 = (_Float16*)take((size_t)3 * 16384 * 2);
  _Float16* wai_16 = (_Float16*)take((size_t)3 * 49152 * 2);
  _Float16* wao_16 = (_Float16*)take((size_t)3 * 16384 * 2);
  _Float16* wm1_16 = (_Float16*)take((size_t)3 * 32768 * 2);
  _Float16* wm2_16 = (_Float16*)take((size_t)3 * 32768 * 2);

  auto cvt = [&](const float* src, _Float16* dst, int n) {
    cvt_kernel<<<(n + 255) / 256, 256, 0, stream>>>(src, dst, n);
  };
  cvt(gine_w1,    wg1_16, 3 * 16384);
  cvt(gine_w2,    wg2_16, 3 * 16384);
  cvt(attn_in_w,  wai_16, 3 * 49152);
  cvt(attn_out_w, wao_16, 3 * 16384);
  cvt(mlp_w1,     wm1_16, 3 * 32768);
  cvt(mlp_w2,     wm2_16, 3 * 32768);

  node_emb_kernel<<<(int)(NC / 256), 256, 0, stream>>>(x, node_emb_w, node_emb_b,
                                                       h32, h16);

  const int MBLK = N_NODES / 128;  // 256 row-blocks per GEMM
  for (int l = 0; l < L_LAYERS; ++l) {
    const float* bg0 = norm_gamma + (size_t)(l * 3 + 0) * C_CH;
    const float* bb0 = norm_beta  + (size_t)(l * 3 + 0) * C_CH;
    const float* bm0 = norm_mean  + (size_t)(l * 3 + 0) * C_CH;
    const float* bv0 = norm_var   + (size_t)(l * 3 + 0) * C_CH;
    const float* bg1 = norm_gamma + (size_t)(l * 3 + 1) * C_CH;
    const float* bb1 = norm_beta  + (size_t)(l * 3 + 1) * C_CH;
    const float* bm1 = norm_mean  + (size_t)(l * 3 + 1) * C_CH;
    const float* bv1 = norm_var   + (size_t)(l * 3 + 1) * C_CH;
    const float* bg2 = norm_gamma + (size_t)(l * 3 + 2) * C_CH;
    const float* bb2 = norm_beta  + (size_t)(l * 3 + 2) * C_CH;
    const float* bm2 = norm_mean  + (size_t)(l * 3 + 2) * C_CH;
    const float* bv2 = norm_var   + (size_t)(l * 3 + 2) * C_CH;

    // --- local branch: GINE
    zero_kernel<<<(int)(NC / 256), 256, 0, stream>>>(agg, (int)NC);
    gine_msg_kernel<<<(E_EDGES * 32) / 256, 256, 0, stream>>>(
        h32, edge_index, edge_attr, lin_edge_w + (size_t)l * C_CH,
        lin_edge_b + (size_t)l * C_CH, agg);
    addcvt_kernel<<<(int)(NC / 256), 256, 0, stream>>>(h32, agg, loc32, loc16,
                                                       (int)NC);
    // t = relu(loc @ gine_w1^T + b1)
    gemm_wmma_kernel<<<dim3(2, MBLK), 128, 0, stream>>>(
        loc16, wg1_16 + (size_t)l * 16384, gine_b1 + (size_t)l * C_CH,
        nullptr, nullptr, nullptr, nullptr, nullptr, nullptr, t16, 128, 128, 1);
    // loc = BN0(t @ gine_w2^T + b2 + h)
    gemm_wmma_kernel<<<dim3(2, MBLK), 128, 0, stream>>>(
        t16, wg2_16 + (size_t)l * 16384, gine_b2 + (size_t)l * C_CH,
        h32, bg0, bb0, bm0, bv0, loc32, nullptr, 128, 128, 0);

    // --- global branch: attention
    gemm_wmma_kernel<<<dim3(6, MBLK), 128, 0, stream>>>(
        h16, wai_16 + (size_t)l * 49152, attn_in_b + (size_t)l * 384,
        nullptr, nullptr, nullptr, nullptr, nullptr, nullptr, qkv16, 128, 384, 0);
    attn_kernel<<<G_GRAPHS * H_HEADS * (NG / 16), 256, 0, stream>>>(qkv16, o16);
    // att = BN1(o @ attn_out_w^T + b + h)
    gemm_wmma_kernel<<<dim3(2, MBLK), 128, 0, stream>>>(
        o16, wao_16 + (size_t)l * 16384, attn_out_b + (size_t)l * C_CH,
        h32, bg1, bb1, bm1, bv1, att32, nullptr, 128, 128, 0);

    // --- combine + FFN
    addcvt_kernel<<<(int)(NC / 256), 256, 0, stream>>>(loc32, att32, comb32,
                                                       comb16, (int)NC);
    gemm_wmma_kernel<<<dim3(4, MBLK), 128, 0, stream>>>(
        comb16, wm1_16 + (size_t)l * 32768, mlp_b1 + (size_t)l * 256,
        nullptr, nullptr, nullptr, nullptr, nullptr, nullptr, t16, 128, 256, 1);
    // h = BN2(t @ mlp_w2^T + b2 + comb)
    gemm_wmma_kernel<<<dim3(2, MBLK), 128, 0, stream>>>(
        t16, wm2_16 + (size_t)l * 32768, mlp_b2 + (size_t)l * C_CH,
        comb32, bg2, bb2, bm2, bv2, h32, h16, 256, 128, 0);
  }

  pool_kernel<<<G_GRAPHS, 128, 0, stream>>>(h32, gpool);
  head_kernel<<<G_GRAPHS, 64, 0, stream>>>(gpool, head_w1, head_b1, head_w2,
                                           head_b2, head_w3, head_b3, out);
}
